// MultiDconvHeadAttention_89936615178404
// MI455X (gfx1250) — matrix operand order repaired and checked by reading, not compiled, not run on hardware
//
#include <hip/hip_runtime.h>
#include <hip/hip_bf16.h>

// MultiDconvHeadAttention for MI455X (gfx1250, wave32, WMMA).
//  1) qkv 1x1 conv as f16 WMMA GEMM (576x192 * 192x4096, f32 accum + bias)
//  2) 3x3 depthwise conv (f32 VALU)
//  3) normalize q/k (torch eps clamp), fold temperature into q, pack f16 layouts
//  4) flash attention, transposed formulation: S^T = K*Q^T, O^T = V^T*P^T.
//     One wave owns 32 query columns; lane = query -> softmax is per-lane
//     reduction + one shfl_xor(16). 14 wmmas per 32-key step, K/V fragments
//     reused across both query column tiles.
//  5) proj 1x1 conv as f16 WMMA GEMM (192x192 * 192x4096, f32 accum + bias)

typedef __attribute__((ext_vector_type(16))) _Float16 v16h;
typedef __attribute__((ext_vector_type(8)))  _Float16 v8h;
typedef __attribute__((ext_vector_type(8)))  float    v8f;

union V16U { v16h v; v8h h[2]; };

__device__ __forceinline__ v8f wmma_f16(v16h a, v16h b, v8f c) {
  return __builtin_amdgcn_wmma_f32_16x16x32_f16(false, a, false, b, (short)0, c,
                                                false, false);
}

// A-matrix fragment (16x32 f16, row-major source [M][stride]):
// lane holds row m = lane&15; base16 = (lane>=16) ? 16 : 0
// halves h -> k = kbase + base16 + h
__device__ __forceinline__ v16h load_a_frag(const _Float16* __restrict__ A, int row,
                                            int stride, int kbase, int lane) {
  const int base16 = (lane & 16) ? 16 : 0;
  const _Float16* p = A + (size_t)row * stride + kbase + base16;
  V16U u;
  u.h[0] = *(const v8h*)(p);
  u.h[1] = *(const v8h*)(p + 8);
  return u.v;
}

// B-matrix fragment (32x16 f16) loaded from Bt stored as [N][stride] (i.e. B^T):
// lane holds col n = lane&15; halves h -> k = kbase + base16 + h
__device__ __forceinline__ v16h load_b_frag(const _Float16* __restrict__ Bt, int col,
                                            int stride, int kbase, int lane) {
  const int base16 = (lane & 16) ? 16 : 0;
  const _Float16* p = Bt + (size_t)col * stride + kbase + base16;
  V16U u;
  u.h[0] = *(const v8h*)(p);
  u.h[1] = *(const v8h*)(p + 8);
  return u.v;
}

// ---------------------------------------------------------------- converts ---
__global__ void k_f32_to_f16(const float* __restrict__ in, _Float16* __restrict__ out,
                             int n) {
  int i = blockIdx.x * blockDim.x + threadIdx.x;
  if (i < n) out[i] = (_Float16)in[i];
}

// x[c][n] (192 x 4096) -> Xt[n][c] f16 (4096 x 192)
__global__ void k_x_transpose(const float* __restrict__ x, _Float16* __restrict__ xt) {
  int i = blockIdx.x * blockDim.x + threadIdx.x;   // c*4096 + n
  int c = i >> 12, n = i & 4095;
  xt[(size_t)n * 192 + c] = (_Float16)x[i];
}

// ---------------------------------------------------------- generic WMMA GEMM
// C[M][N] = A(f16,[M][K]) * Bt(f16,[N][K])^T + bias[M]
// One wave per 16x64 tile: A-fragment reused across 4 N-tiles.
__global__ __launch_bounds__(32) void k_gemm_wmma(const _Float16* __restrict__ A,
                                                  const _Float16* __restrict__ Bt,
                                                  const float* __restrict__ bias,
                                                  float* __restrict__ C,
                                                  int N, int K) {
  const int mt = blockIdx.x, nt4 = blockIdx.y;
  const int lane = threadIdx.x;
  const int row = mt * 16 + (lane & 15);
  v8f acc[4] = {};
  for (int kb = 0; kb < K; kb += 32) {
    v16h a = load_a_frag(A, row, K, kb, lane);
#pragma unroll
    for (int t = 0; t < 4; ++t) {
      v16h b = load_b_frag(Bt, nt4 * 64 + t * 16 + (lane & 15), K, kb, lane);
      acc[t] = wmma_f16(a, b, acc[t]);
    }
  }
  const int rbase = mt * 16 + ((lane & 16) ? 8 : 0);
#pragma unroll
  for (int t = 0; t < 4; ++t) {
    int n = nt4 * 64 + t * 16 + (lane & 15);
#pragma unroll
    for (int r = 0; r < 8; ++r) {
      int m = rbase + r;
      C[(size_t)m * N + n] = acc[t][r] + bias[m];
    }
  }
}

// ------------------------------------------------------------ depthwise 3x3 --
__global__ void k_dwconv3x3(const float* __restrict__ in, const float* __restrict__ w,
                            const float* __restrict__ b, float* __restrict__ out) {
  const int ch = blockIdx.y;
  const int p = blockIdx.x * blockDim.x + threadIdx.x;   // 0..4095
  const int y = p >> 6, x = p & 63;
  const float* wp = w + ch * 9;
  const float* ip = in + (size_t)ch * 4096;
  float acc = b[ch];
#pragma unroll
  for (int ky = 0; ky < 3; ++ky) {
    int yy = y + ky - 1;
    if (yy < 0 || yy > 63) continue;
#pragma unroll
    for (int kx = 0; kx < 3; ++kx) {
      int xx = x + kx - 1;
      if (xx < 0 || xx > 63) continue;
      acc += ip[yy * 64 + xx] * wp[ky * 3 + kx];
    }
  }
  out[(size_t)ch * 4096 + p] = acc;
}

// --------------------------------------------- normalize q,k + pack f16 -----
// dw: [576][4096] f32 (q:0..191, k:192..383, v:384..575)
// Qh gets temperature folded in. Qh/Kh: [head][n][64] f16 (d padded 48->64),
// Vt: [head][d=48][n] f16 (transposed, serves as A-fragments for O^T=V^T*P^T).
__global__ void k_pack_qkv(const float* __restrict__ dw, const float* __restrict__ temp,
                           _Float16* __restrict__ Qh, _Float16* __restrict__ Kh,
                           _Float16* __restrict__ Vt) {
  const int n = blockIdx.x * blockDim.x + threadIdx.x;   // 0..4095
  const int head = blockIdx.y;                           // 0..3
  const float* qp = dw + (size_t)(head * 48) * 4096;
  const float* kp = dw + (size_t)(192 + head * 48) * 4096;
  const float* vp = dw + (size_t)(384 + head * 48) * 4096;
  float qn = 0.f, kn = 0.f;
  for (int d = 0; d < 48; ++d) {
    float q = qp[(size_t)d * 4096 + n]; qn += q * q;
    float k = kp[(size_t)d * 4096 + n]; kn += k * k;
  }
  qn = temp[head] / fmaxf(sqrtf(qn), 1e-12f);   // fold temperature into q
  kn = 1.0f / fmaxf(sqrtf(kn), 1e-12f);
  _Float16* qrow = Qh + ((size_t)head * 4096 + n) * 64;
  _Float16* krow = Kh + ((size_t)head * 4096 + n) * 64;
  for (int d = 0; d < 48; ++d) {
    qrow[d] = (_Float16)(qp[(size_t)d * 4096 + n] * qn);
    krow[d] = (_Float16)(kp[(size_t)d * 4096 + n] * kn);
    Vt[((size_t)head * 48 + d) * 4096 + n] = (_Float16)vp[(size_t)d * 4096 + n];
  }
  for (int d = 48; d < 64; ++d) { qrow[d] = (_Float16)0.f; krow[d] = (_Float16)0.f; }
}

// ------------------------------------------------------------ flash attention
// Transposed: S^T = K*Q^T (lane = query column), O^T = V^T * P^T.
// One wave owns 32 queries (2 column tiles); loop over keys in tiles of 32.
// Per step: 8 wmmas for S^T + 6 wmmas for O^T, K/V fragments shared by both
// column tiles. Ot: [n][192] f16 (c = head*48+d), B^T-ready for the proj GEMM.
__global__ __launch_bounds__(32) void k_flash_attn(const _Float16* __restrict__ Qh,
                                                   const _Float16* __restrict__ Kh,
                                                   const _Float16* __restrict__ Vt,
                                                   _Float16* __restrict__ Ot) {
  __shared__ __align__(16) _Float16 lds_pT[2 * 16 * 32];   // P^T staging, 2 KB
  const int lane = threadIdx.x;
  const int qt = blockIdx.x;          // 0..127 (32 queries each)
  const int head = blockIdx.y;        // 0..3
  const _Float16* Q = Qh + (size_t)head * 4096 * 64;
  const _Float16* K = Kh + (size_t)head * 4096 * 64;
  const _Float16* V = Vt + (size_t)head * 48 * 4096;
  const int colh = lane & 15;
  const int off8 = (lane & 16) ? 8 : 0;
  const int base16 = (lane & 16) ? 16 : 0;
  const int q0 = qt * 32;

  // Q as B-fragments, resident in registers for the whole key loop
  v16h bq[2][2];
#pragma unroll
  for (int ct = 0; ct < 2; ++ct) {
    bq[ct][0] = load_b_frag(Q, q0 + ct * 16 + colh, 64, 0, lane);
    bq[ct][1] = load_b_frag(Q, q0 + ct * 16 + colh, 64, 32, lane);
  }

  v8f ot[2][3] = {};                 // O^T accum: [col tile][16-dim tile]
  float mq[2] = {-1e30f, -1e30f};    // running max per owned query
  float lq[2] = {0.f, 0.f};          // running sum per owned query

  for (int j = 0; j < 4096; j += 32) {
    // K rows as A-fragments (shared across both query column tiles)
    v16h ka[2][2];
#pragma unroll
    for (int mt = 0; mt < 2; ++mt) {
      ka[mt][0] = load_a_frag(K, j + mt * 16 + colh, 64, 0, lane);
      ka[mt][1] = load_a_frag(K, j + mt * 16 + colh, 64, 32, lane);
    }

#pragma unroll
    for (int ct = 0; ct < 2; ++ct) {
      // S^T tiles: rows = keys, col = this lane's query
      v8f s0 = {}, s1 = {};
      s0 = wmma_f16(ka[0][0], bq[ct][0], s0);
      s0 = wmma_f16(ka[0][1], bq[ct][1], s0);
      s1 = wmma_f16(ka[1][0], bq[ct][0], s1);
      s1 = wmma_f16(ka[1][1], bq[ct][1], s1);

      // online softmax: per-lane reduction over 16 scores + one pair-merge
      float mx = s0[0];
#pragma unroll
      for (int r = 1; r < 8; ++r) mx = fmaxf(mx, s0[r]);
#pragma unroll
      for (int r = 0; r < 8; ++r) mx = fmaxf(mx, s1[r]);
      mx = fmaxf(mx, __shfl_xor(mx, 16, 32));
      float mnew = fmaxf(mq[ct], mx);
      float sc = __expf(mq[ct] - mnew);
      mq[ct] = mnew;

      float rs = 0.f;
      _Float16* lp = lds_pT + ct * 512 + colh * 32;
#pragma unroll
      for (int r = 0; r < 8; ++r) {
        float p0 = __expf(s0[r] - mnew);
        float p1 = __expf(s1[r] - mnew);
        rs += p0 + p1;
        lp[r + off8]      = (_Float16)p0;   // keys j+0..15
        lp[16 + r + off8] = (_Float16)p1;   // keys j+16..31
      }
      rs += __shfl_xor(rs, 16, 32);
      lq[ct] = lq[ct] * sc + rs;
#pragma unroll
      for (int dt = 0; dt < 3; ++dt)
#pragma unroll
        for (int r = 0; r < 8; ++r) ot[ct][dt][r] *= sc;
    }
    __syncthreads();   // single wave: needed dscnt ordering for LDS RAW

    // V^T rows as A-fragments (48 dims = 3 tiles, K = 32 keys at offset j)
    v16h va[3];
#pragma unroll
    for (int dt = 0; dt < 3; ++dt)
      va[dt] = load_a_frag(V, dt * 16 + colh, 4096, j, lane);

#pragma unroll
    for (int ct = 0; ct < 2; ++ct) {
      V16U pu;   // P^T as B-fragment: lane's query column, 16 contiguous keys
      pu.h[0] = *(const v8h*)(lds_pT + ct * 512 + colh * 32 + base16);
      pu.h[1] = *(const v8h*)(lds_pT + ct * 512 + colh * 32 + base16 + 8);
#pragma unroll
      for (int dt = 0; dt < 3; ++dt)
        ot[ct][dt] = wmma_f16(va[dt], pu.v, ot[ct][dt]);
    }
    __syncthreads();   // protect lds_pT before next iteration
  }

  // epilogue: normalize by row sums, write Ot[n][c]
#pragma unroll
  for (int ct = 0; ct < 2; ++ct) {
    float inv = 1.0f / lq[ct];
    int n = q0 + ct * 16 + colh;
    _Float16* orow = Ot + (size_t)n * 192 + head * 48;
#pragma unroll
    for (int dt = 0; dt < 3; ++dt)
#pragma unroll
      for (int r = 0; r < 8; ++r)
        orow[dt * 16 + r + off8] = (_Float16)(ot[ct][dt][r] * inv);
  }
}

// ----------------------------------------------------------------- launcher --
extern "C" void kernel_launch(void* const* d_in, const int* in_sizes, int n_in,
                              void* d_out, int out_size, void* d_ws, size_t ws_size,
                              hipStream_t stream) {
  const float* x      = (const float*)d_in[0];   // (1,192,64,64)
  const float* w_qkv  = (const float*)d_in[1];   // (576,192)
  const float* b_qkv  = (const float*)d_in[2];   // (576,)
  const float* w_dw   = (const float*)d_in[3];   // (576,1,3,3)
  const float* b_dw   = (const float*)d_in[4];   // (576,)
  const float* w_proj = (const float*)d_in[5];   // (192,192)
  const float* b_proj = (const float*)d_in[6];   // (192,)
  const float* temp   = (const float*)d_in[7];   // (4,1,1)
  float* out = (float*)d_out;                    // (1,192,64,64)

  char* ws = (char*)d_ws;
  size_t off = 0;
  _Float16* Wqkv_h = (_Float16*)(ws + off); off += (size_t)576 * 192 * 2;
  _Float16* Wprj_h = (_Float16*)(ws + off); off += (size_t)192 * 192 * 2;
  _Float16* Xt_h   = (_Float16*)(ws + off); off += (size_t)4096 * 192 * 2;
  float*    qkv_f  = (float*)   (ws + off); off += (size_t)576 * 4096 * 4;
  float*    dw_f   = (float*)   (ws + off); off += (size_t)576 * 4096 * 4;
  _Float16* Qh     = (_Float16*)(ws + off); off += (size_t)4 * 4096 * 64 * 2;
  _Float16* Kh     = (_Float16*)(ws + off); off += (size_t)4 * 4096 * 64 * 2;
  _Float16* Vt     = (_Float16*)(ws + off); off += (size_t)4 * 48 * 4096 * 2;
  _Float16* Ot_h   = (_Float16*)(ws + off); off += (size_t)4096 * 192 * 2;
  (void)ws_size; (void)in_sizes; (void)n_in; (void)out_size;

  // 1) weight + activation converts
  k_f32_to_f16<<<(576 * 192 + 255) / 256, 256, 0, stream>>>(w_qkv, Wqkv_h, 576 * 192);
  k_f32_to_f16<<<(192 * 192 + 255) / 256, 256, 0, stream>>>(w_proj, Wprj_h, 192 * 192);
  k_x_transpose<<<(192 * 4096) / 256, 256, 0, stream>>>(x, Xt_h);

  // 2) qkv = Wqkv * X  (576 x 4096), WMMA, 16x64 tiles per wave
  k_gemm_wmma<<<dim3(576 / 16, 4096 / 64), 32, 0, stream>>>(Wqkv_h, Xt_h, b_qkv,
                                                            qkv_f, 4096, 192);

  // 3) 3x3 depthwise conv
  k_dwconv3x3<<<dim3(4096 / 256, 576), 256, 0, stream>>>(qkv_f, w_dw, b_dw, dw_f);

  // 4) normalize (+temperature fold) + pack f16 fragment layouts
  k_pack_qkv<<<dim3(4096 / 256, 4), 256, 0, stream>>>(dw_f, temp, Qh, Kh, Vt);

  // 5) flash attention (WMMA), one wave per 32-query tile per head
  k_flash_attn<<<dim3(4096 / 32, 4), 32, 0, stream>>>(Qh, Kh, Vt, Ot_h);

  // 6) out = Wproj * attn_out (192 x 4096), WMMA, writes final NCHW output
  k_gemm_wmma<<<dim3(192 / 16, 4096 / 64), 32, 0, stream>>>(Wprj_h, Ot_h, b_proj,
                                                            out, 4096, 192);
}
